// TSBlock_32323923869717
// MI455X (gfx1250) — compile-verified, matrix operand order checked by reading, not run on hardware
//
#include <hip/hip_runtime.h>
#include <hip/hip_bf16.h>

// ---------------------------------------------------------------------------
// Transformer block for MI455X (gfx1250): bf16 WMMA everywhere, f32 accum,
// TDM (tensor_load_to_lds) double-buffered A-tile staging in the GEMMs.
// D=1024, H=16, hd=64, T=2048, B=2, F=4096, MAX_DIST=512.
// ---------------------------------------------------------------------------

typedef __bf16 bf16;
typedef __attribute__((ext_vector_type(16))) __bf16 v16bf;
typedef __attribute__((ext_vector_type(8)))  float  v8f;
typedef __attribute__((ext_vector_type(4)))  unsigned int u32x4;
typedef __attribute__((ext_vector_type(8)))  int i32x8;
typedef __attribute__((ext_vector_type(4)))  int i32x4;

#define D_MODEL 1024
#define N_HEADS 16
#define HEAD_D  64
#define SEQ_T   2048
#define BATCH   2
#define FF_DIM  4096
#define ROWS_M  (BATCH * SEQ_T)   // 4096

#if __has_builtin(__builtin_amdgcn_tensor_load_to_lds)
#define HAVE_TDM 1
#else
#define HAVE_TDM 0
#endif

// ---------------------------------------------------------------------------
// Fragment loaders.  16-bit A/B fragment layout (ISA 7.12.2):
//   lane<16 : row/col = lane,    K = {0..7} then {16..23}
//   lane>=16: row/col = lane-16, K = {8..15} then {24..31}
// Per lane: two contiguous 16B reads at +kb and +kb+16 elems (kb = 0 or 8).
// ---------------------------------------------------------------------------
__device__ __forceinline__ v16bf load_frag(const bf16* tile, int ldt) {
  int lane = threadIdx.x & 31;
  const bf16* p = tile + (lane & 15) * ldt + ((lane >> 4) << 3);
  union { uint4 u[2]; v16bf v; } x;
  x.u[0] = *(const uint4*)(p);
  x.u[1] = *(const uint4*)(p + 16);
  return x.v;
}

__device__ __forceinline__ v16bf load_frag_g(const bf16* p) {
  union { uint4 u[2]; v16bf v; } x;
  x.u[0] = *(const uint4*)(p);
  x.u[1] = *(const uint4*)(p + 16);
  return x.v;
}

__device__ __forceinline__ v8f wmma_bf16(v16bf a, v16bf b, v8f c) {
  return __builtin_amdgcn_wmma_f32_16x16x32_bf16(false, a, false, b,
                                                 (short)0, c, false, false);
}

__device__ __forceinline__ float gelu_exact(float x) {
  return 0.5f * x * (1.0f + erff(x * 0.70710678118654752f));
}

// ---------------------------------------------------------------------------
// f32 -> bf16 conversion
// ---------------------------------------------------------------------------
__global__ __launch_bounds__(256) void f2b_kernel(const float* __restrict__ in,
                                                  bf16* __restrict__ out, int n) {
  int i = blockIdx.x * 256 + threadIdx.x;
  if (i < n) out[i] = (bf16)in[i];
}

// ---------------------------------------------------------------------------
// Tiled bf16 GEMM: C[M,N] = A[M,K] @ B[K,N], fused epilogues.
// Block tile 128x128x32, 256 threads = 8 waves (2x4), 4x2 WMMA tiles/wave.
// A tile staged by the Tensor Data Mover with HW row padding (64B rows +
// 16B pad -> 40-element LDS stride), double buffered, synced on TENSORcnt.
// ---------------------------------------------------------------------------
#define BM 128
#define BN 128
#define BK 32
#define LDS_STRIDE 40   // BK + 8 pad elements (80B rows, 16B-aligned)

#if HAVE_TDM
// D# for a 2D bf16 tile copy: tile_dim0=32 elems (64B rows), tile_dim1=128,
// tensor_dim0_stride = Kdim, LDS pad: every 16 dwords (64B) insert 4 dwords.
__device__ __forceinline__ void tdm_load_a_tile(const bf16* gsrc, void* lds_dst,
                                                int Kdim, int Mdim) {
  unsigned long long ga = (unsigned long long)(uintptr_t)gsrc;
  unsigned int lds = (unsigned int)(uintptr_t)lds_dst;  // flat->LDS truncation
  u32x4 g0;
  g0.x = 1u;                                            // count = 1 descriptor
  g0.y = lds;                                           // lds_addr
  g0.z = (unsigned int)ga;                              // global_addr[31:0]
  g0.w = (unsigned int)((ga >> 32) & 0x01FFFFFFu)       // global_addr[56:32]
         | 0x80000000u;                                 // type = 2 ("image")
  i32x8 g1;
  g1[0] = (int)((1u << 16)      // data_size = 1 (2 bytes)
              | (1u << 20)      // pad_enable
              | (3u << 22)      // pad_interval: 16 dwords (64B) per row
              | (3u << 25));    // pad_amount: 4 dwords (16B)
  g1[1] = (int)(((unsigned)Kdim & 0xFFFFu) << 16);               // tensor_dim0 lo
  g1[2] = (int)(((unsigned)Kdim >> 16)
              | (((unsigned)Mdim & 0xFFFFu) << 16));             // dim0 hi|dim1 lo
  g1[3] = (int)(((unsigned)Mdim >> 16) | ((unsigned)BK << 16));  // dim1 hi|tile0
  g1[4] = (int)BM;                                               // tile1 | tile2=0
  g1[5] = (int)Kdim;                                             // dim0_stride lo32
  g1[6] = 0;                                                     // stride hi16
  g1[7] = 0;
  i32x4 z4 = {0, 0, 0, 0};
#if __clang_major__ >= 23
  i32x8 z8 = {};
  __builtin_amdgcn_tensor_load_to_lds(g0, g1, z4, z4, z8, 0);
#else
  __builtin_amdgcn_tensor_load_to_lds(g0, g1, z4, z4, 0);
#endif
}
#endif

template <int MODE>
__global__ __launch_bounds__(256) void gemm_bf16_kernel(
    const bf16* __restrict__ A, const bf16* __restrict__ Bm,
    int Mdim, int Ndim, int Kdim,
    const float* __restrict__ bias, const float* __restrict__ res,
    float* __restrict__ outf, bf16* __restrict__ outb,
    bf16* __restrict__ qout, bf16* __restrict__ kout, bf16* __restrict__ vtout) {
  __shared__ bf16 As[2][BM][LDS_STRIDE];
  __shared__ bf16 Bs[BN][LDS_STRIDE];   // B tile stored transposed: Bs[n][k]

  const int n0 = blockIdx.x * BN;
  const int m0 = blockIdx.y * BM;
  const int tid = threadIdx.x;
  const int lane = tid & 31;
  const int wid = tid >> 5;
  const int wm = (wid & 1) * 64;   // wave row offset inside block tile
  const int wn = (wid >> 1) * 32;  // wave col offset inside block tile

  v8f acc[4][2];
#pragma unroll
  for (int i = 0; i < 4; ++i)
#pragma unroll
    for (int j = 0; j < 2; ++j) acc[i][j] = (v8f){};

  const int ar = tid >> 2;          // 0..63   (fallback A staging)
  const int ac = (tid & 3) * 8;     // 0,8,16,24

#if HAVE_TDM
  const bool tdm_wave = (wid == 0);
  if (tdm_wave) tdm_load_a_tile(&A[(size_t)m0 * Kdim], &As[0][0][0], Kdim, Mdim);
#endif

  int cur = 0;
  for (int k0 = 0; k0 < Kdim; k0 += BK, cur ^= 1) {
#if !HAVE_TDM
    // ---- fallback: stage A tile through VGPRs ----
#pragma unroll
    for (int c = 0; c < 2; ++c) {
      int row = ar + c * 64;
      *(uint4*)&As[cur][row][ac] =
          *(const uint4*)&A[(size_t)(m0 + row) * Kdim + k0 + ac];
    }
#endif
    // ---- stage B tile transposed ----
#pragma unroll
    for (int c = 0; c < 2; ++c) {
      int g = tid + c * 256;        // 512 groups of 8 elems
      int kk = g >> 4;              // 0..31
      int nn = (g & 15) * 8;        // 0..120
      union { uint4 u; bf16 e[8]; } t;
      t.u = *(const uint4*)&Bm[(size_t)(k0 + kk) * Ndim + n0 + nn];
#pragma unroll
      for (int i = 0; i < 8; ++i) Bs[nn + i][kk] = t.e[i];
    }
    if (k0 + BK < Kdim)   // prefetch next B tile (global_prefetch_b8)
      __builtin_prefetch(&Bm[(size_t)(k0 + BK + (tid >> 4)) * Ndim + n0], 0, 1);
#if HAVE_TDM
    if (tdm_wave) __builtin_amdgcn_s_wait_tensorcnt(0);  // A(cur) landed
#endif
    __syncthreads();
#if HAVE_TDM
    if (tdm_wave && (k0 + BK < Kdim))  // kick DMA for next A tile, overlap WMMA
      tdm_load_a_tile(&A[(size_t)m0 * Kdim + (k0 + BK)], &As[cur ^ 1][0][0],
                      Kdim, Mdim);
#endif

    v16bf af[4], bfr[2];
#pragma unroll
    for (int i = 0; i < 4; ++i)
      af[i] = load_frag(&As[cur][wm + i * 16][0], LDS_STRIDE);
#pragma unroll
    for (int j = 0; j < 2; ++j) bfr[j] = load_frag(&Bs[wn + j * 16][0], LDS_STRIDE);
#pragma unroll
    for (int i = 0; i < 4; ++i)
#pragma unroll
      for (int j = 0; j < 2; ++j) acc[i][j] = wmma_bf16(af[i], bfr[j], acc[i][j]);
    __syncthreads();
  }

  // ---- epilogue: C layout: VGPR r -> M = (lane>=16 ? 8 : 0)+r, N = lane&15
  const int nl = lane & 15;
  const int mb = (lane >> 4) * 8;
#pragma unroll
  for (int i = 0; i < 4; ++i) {
#pragma unroll
    for (int j = 0; j < 2; ++j) {
#pragma unroll
      for (int r = 0; r < 8; ++r) {
        int gm = m0 + wm + i * 16 + mb + r;
        int gn = n0 + wn + j * 16 + nl;
        float v = acc[i][j][r] + bias[gn];
        if constexpr (MODE == 0) {
          int b = gm >> 11, t = gm & 2047;
          int which = gn >> 10, dfull = gn & 1023;
          int h = dfull >> 6, d = dfull & 63;
          size_t bh = (size_t)(b * N_HEADS + h);
          if (which == 0)      qout[(bh * SEQ_T + t) * HEAD_D + d] = (bf16)v;
          else if (which == 1) kout[(bh * SEQ_T + t) * HEAD_D + d] = (bf16)v;
          else                 vtout[(bh * HEAD_D + d) * SEQ_T + t] = (bf16)v;
        } else if constexpr (MODE == 1 || MODE == 3) {
          v += res[(size_t)gm * Ndim + gn];
          outf[(size_t)gm * Ndim + gn] = v;
        } else {  // MODE 2: GELU -> bf16
          outb[(size_t)gm * Ndim + gn] = (bf16)gelu_exact(v);
        }
      }
    }
  }
}

// ---------------------------------------------------------------------------
// Fused attention: one workgroup per (b, h, 32-query tile).
// LDS: S[32][2056] f32 (263168B) + Qs[32][72] bf16 (4608B) = 267776B (<320KB).
// ---------------------------------------------------------------------------
#define SROW 2056                     // f32 stride per S row (pad 8)
#define PROW (SROW * 2)               // same row viewed as bf16 elements
#define QROW 72
#define ATTN_LDS_BYTES (32 * SROW * 4 + 32 * QROW * 2)

__global__ __launch_bounds__(256) void attn_kernel(
    const bf16* __restrict__ Qb, const bf16* __restrict__ Kb,
    const bf16* __restrict__ Vtb, const float* __restrict__ rel_bias,
    float* __restrict__ attn_out, bf16* __restrict__ AoB) {
  extern __shared__ char smem[];
  float* Sf = (float*)smem;
  bf16* Pb = (bf16*)smem;                       // bf16 alias of S rows
  bf16* Qs = (bf16*)(smem + 32 * SROW * 4);

  const int q0 = blockIdx.x * 32;
  const int h = blockIdx.y;
  const int b = blockIdx.z;
  const int tid = threadIdx.x;
  const int lane = tid & 31;
  const int wid = tid >> 5;
  const size_t bh = (size_t)(b * N_HEADS + h);

  // ---- stage 32x64 Q tile ----
  {
    int row = tid >> 3, col = (tid & 7) * 8;
    *(uint4*)&Qs[row * QROW + col] =
        *(const uint4*)&Qb[(bh * SEQ_T + q0 + row) * HEAD_D + col];
  }
  __syncthreads();

  // ---- phase 1: scores.  wave 'wid' covers key cols [wid*256, wid*256+256)
  const float scale = 0.125f;  // 1/sqrt(64)
  const int nl = lane & 15;
  const int mb = (lane >> 4) * 8;
  for (int nf = 0; nf < 16; ++nf) {
    int n0 = wid * 256 + nf * 16;
    v8f acc0 = (v8f){}, acc1 = (v8f){};
#pragma unroll
    for (int kc = 0; kc < HEAD_D; kc += 32) {
      v16bf a0 = load_frag(&Qs[0 * QROW + kc], QROW);
      v16bf a1 = load_frag(&Qs[16 * QROW + kc], QROW);
      const bf16* kp = &Kb[(bh * SEQ_T + n0 + nl) * HEAD_D + kc + ((lane >> 4) << 3)];
      v16bf bb = load_frag_g(kp);
      acc0 = wmma_bf16(a0, bb, acc0);
      acc1 = wmma_bf16(a1, bb, acc1);
    }
    int jn = n0 + nl;
#pragma unroll
    for (int r = 0; r < 8; ++r) {
      int m = mb + r;
      int rel0 = jn - (q0 + m);
      rel0 = rel0 < -512 ? -512 : (rel0 > 512 ? 512 : rel0);
      Sf[m * SROW + jn] = acc0[r] * scale + rel_bias[(rel0 + 512) * N_HEADS + h];
      int m1 = m + 16;
      int rel1 = jn - (q0 + m1);
      rel1 = rel1 < -512 ? -512 : (rel1 > 512 ? 512 : rel1);
      Sf[m1 * SROW + jn] = acc1[r] * scale + rel_bias[(rel1 + 512) * N_HEADS + h];
    }
  }
  __syncthreads();

  // ---- phase 2: softmax, wave per row (rows 4*wid .. 4*wid+3)
  for (int rr = 0; rr < 4; ++rr) {
    int row = wid * 4 + rr;
    volatile float* Srow = &Sf[row * SROW];
    float mx = -3.4e38f;
    for (int j = lane; j < SEQ_T; j += 32) mx = fmaxf(mx, Srow[j]);
#pragma unroll
    for (int o = 16; o > 0; o >>= 1) mx = fmaxf(mx, __shfl_xor(mx, o, 32));
    float sum = 0.f;
    for (int j = lane; j < SEQ_T; j += 32) sum += __expf(Srow[j] - mx);
#pragma unroll
    for (int o = 16; o > 0; o >>= 1) sum += __shfl_xor(sum, o, 32);
    float inv = 1.0f / sum;
    float* orow = attn_out + (((size_t)bh * SEQ_T + q0 + row) << 11);
    volatile bf16* Prow = &Pb[row * PROW];
    for (int j = lane; j < SEQ_T; j += 32) {
      float p = __expf(Srow[j] - mx) * inv;
      __builtin_nontemporal_store(p, &orow[j]);
      // in-place bf16: byte 2j lies inside f32 slot j/2, consumed in a
      // strictly earlier (or same, already-loaded) iteration -> safe.
      Prow[j] = (bf16)p;
    }
  }
  __syncthreads();

  // ---- phase 3: O = P @ V.  8 waves = 2 (m) x 4 (n) tiles, full K=2048.
  const int mf = (wid & 1) * 16;
  const int nf = (wid >> 1) * 16;
  v8f oacc = (v8f){};
  for (int kc = 0; kc < SEQ_T; kc += 32) {
    v16bf a = load_frag(&Pb[mf * PROW + kc], PROW);
    const bf16* vp = &Vtb[(bh * HEAD_D + nf + nl) * SEQ_T + kc + ((lane >> 4) << 3)];
    v16bf bb = load_frag_g(vp);
    oacc = wmma_bf16(a, bb, oacc);
  }
#pragma unroll
  for (int r = 0; r < 8; ++r) {
    int m = mf + mb + r;  // 0..31
    int n = nf + nl;      // 0..63
    AoB[((size_t)b * SEQ_T + q0 + m) * D_MODEL + h * HEAD_D + n] = (bf16)oacc[r];
  }
}

// ---------------------------------------------------------------------------
// Row LayerNorm over D=1024: outf = LN(in)*g+beta (f32), outb optional bf16.
// ---------------------------------------------------------------------------
__global__ __launch_bounds__(256) void ln_kernel(
    const float* __restrict__ in, const float* __restrict__ g,
    const float* __restrict__ beta, float* __restrict__ outf,
    bf16* __restrict__ outb) {
  __shared__ float rs[8], rs2[8], stats[2];
  const int row = blockIdx.x;
  const float* xr = in + (size_t)row * D_MODEL;
  const int tid = threadIdx.x, lane = tid & 31, wid = tid >> 5;
  float s = 0.f, s2 = 0.f;
  for (int j = tid; j < D_MODEL; j += 256) {
    float v = xr[j];
    s += v;
    s2 += v * v;
  }
#pragma unroll
  for (int o = 16; o > 0; o >>= 1) {
    s += __shfl_xor(s, o, 32);
    s2 += __shfl_xor(s2, o, 32);
  }
  if (lane == 0) { rs[wid] = s; rs2[wid] = s2; }
  __syncthreads();
  if (tid == 0) {
    float a = 0.f, c = 0.f;
    for (int i = 0; i < 8; ++i) { a += rs[i]; c += rs2[i]; }
    float mu = a * (1.0f / D_MODEL);
    float var = c * (1.0f / D_MODEL) - mu * mu;
    stats[0] = mu;
    stats[1] = rsqrtf(var + 1e-5f);
  }
  __syncthreads();
  float mu = stats[0], rstd = stats[1];
  for (int j = tid; j < D_MODEL; j += 256) {
    float v = (xr[j] - mu) * rstd * g[j] + beta[j];
    if (outf) outf[(size_t)row * D_MODEL + j] = v;
    if (outb) outb[(size_t)row * D_MODEL + j] = (bf16)v;
  }
}

// ---------------------------------------------------------------------------
// Host side
// ---------------------------------------------------------------------------
extern "C" void kernel_launch(void* const* d_in, const int* in_sizes, int n_in,
                              void* d_out, int out_size, void* d_ws, size_t ws_size,
                              hipStream_t stream) {
  const float* x        = (const float*)d_in[0];
  const float* w_qkv    = (const float*)d_in[1];
  const float* b_qkv    = (const float*)d_in[2];
  const float* w_out    = (const float*)d_in[3];
  const float* b_out    = (const float*)d_in[4];
  const float* rel_bias = (const float*)d_in[5];
  const float* g1       = (const float*)d_in[6];
  const float* beta1    = (const float*)d_in[7];
  const float* w1       = (const float*)d_in[8];
  const float* b1       = (const float*)d_in[9];
  const float* w2       = (const float*)d_in[10];
  const float* b2       = (const float*)d_in[11];
  const float* g2       = (const float*)d_in[12];
  const float* beta2    = (const float*)d_in[13];

  float* out_x3   = (float*)d_out;
  float* out_attn = out_x3 + (size_t)ROWS_M * D_MODEL;

  char* ws = (char*)d_ws;
  size_t off = 0;
  auto carve = [&](size_t bytes) {
    void* p = ws + off;
    off += (bytes + 255) & ~(size_t)255;
    return p;
  };
  bf16*  xb    = (bf16*)carve((size_t)ROWS_M * D_MODEL * 2);
  bf16*  wqkvb = (bf16*)carve((size_t)D_MODEL * 3 * D_MODEL * 2);
  bf16*  woutb = (bf16*)carve((size_t)D_MODEL * D_MODEL * 2);
  bf16*  w1b   = (bf16*)carve((size_t)D_MODEL * FF_DIM * 2);
  bf16*  w2b   = (bf16*)carve((size_t)FF_DIM * D_MODEL * 2);
  bf16*  Qb    = (bf16*)carve((size_t)BATCH * N_HEADS * SEQ_T * HEAD_D * 2);
  bf16*  Kb    = (bf16*)carve((size_t)BATCH * N_HEADS * SEQ_T * HEAD_D * 2);
  bf16*  Vtb   = (bf16*)carve((size_t)BATCH * N_HEADS * HEAD_D * SEQ_T * 2);
  bf16*  AoB   = (bf16*)carve((size_t)ROWS_M * D_MODEL * 2);
  bf16*  x2b   = (bf16*)carve((size_t)ROWS_M * D_MODEL * 2);
  bf16*  hb    = (bf16*)carve((size_t)ROWS_M * FF_DIM * 2);
  float* tmp1  = (float*)carve((size_t)ROWS_M * D_MODEL * 4);
  float* x2f   = (float*)carve((size_t)ROWS_M * D_MODEL * 4);

  auto cvt = [&](const float* src, bf16* dst, size_t n) {
    f2b_kernel<<<dim3((unsigned)((n + 255) / 256)), 256, 0, stream>>>(src, dst, (int)n);
  };
  cvt(x,     xb,    (size_t)ROWS_M * D_MODEL);
  cvt(w_qkv, wqkvb, (size_t)D_MODEL * 3 * D_MODEL);
  cvt(w_out, woutb, (size_t)D_MODEL * D_MODEL);
  cvt(w1,    w1b,   (size_t)D_MODEL * FF_DIM);
  cvt(w2,    w2b,   (size_t)FF_DIM * D_MODEL);

  // 1) QKV projection: [4096,1024] @ [1024,3072]
  gemm_bf16_kernel<0><<<dim3(3 * D_MODEL / BN, ROWS_M / BM), 256, 0, stream>>>(
      xb, wqkvb, ROWS_M, 3 * D_MODEL, D_MODEL, b_qkv, nullptr, nullptr, nullptr,
      Qb, Kb, Vtb);

  // 2) fused attention (scores + softmax + attn output + P@V)
  attn_kernel<<<dim3(SEQ_T / 32, N_HEADS, BATCH), 256, ATTN_LDS_BYTES, stream>>>(
      Qb, Kb, Vtb, rel_bias, out_attn, AoB);

  // 3) out projection + residual -> tmp1 (f32)
  gemm_bf16_kernel<1><<<dim3(D_MODEL / BN, ROWS_M / BM), 256, 0, stream>>>(
      AoB, woutb, ROWS_M, D_MODEL, D_MODEL, b_out, x, tmp1, nullptr,
      nullptr, nullptr, nullptr);

  // 4) LN1 -> x2f (f32) + x2b (bf16)
  ln_kernel<<<dim3(ROWS_M), 256, 0, stream>>>(tmp1, g1, beta1, x2f, x2b);

  // 5) FF1 + GELU -> hb (bf16): [4096,1024] @ [1024,4096]
  gemm_bf16_kernel<2><<<dim3(FF_DIM / BN, ROWS_M / BM), 256, 0, stream>>>(
      x2b, w1b, ROWS_M, FF_DIM, D_MODEL, b1, nullptr, nullptr, hb,
      nullptr, nullptr, nullptr);

  // 6) FF2 + residual -> tmp1 (f32): [4096,4096] @ [4096,1024]
  gemm_bf16_kernel<3><<<dim3(D_MODEL / BN, ROWS_M / BM), 256, 0, stream>>>(
      hb, w2b, ROWS_M, D_MODEL, FF_DIM, b2, x2f, tmp1, nullptr,
      nullptr, nullptr, nullptr);

  // 7) LN2 -> x3 (f32, first chunk of d_out)
  ln_kernel<<<dim3(ROWS_M), 256, 0, stream>>>(tmp1, g2, beta2, out_x3, nullptr);
}
